// LSTM_74397423501322
// MI455X (gfx1250) — compile-verified, matrix operand order checked by reading, not run on hardware
//
#include <hip/hip_runtime.h>
#include <math.h>

// ---------------------------------------------------------------------------
// 2-layer LSTM, B=256, T=128, I=256, H=512 — CDNA5 (gfx1250) WMMA bf16 path.
//
// Grid: 16 workgroups x 512 threads (16 waves). Each WG owns a 16-row batch
// tile for all 128 timesteps (batch rows independent -> no inter-WG sync).
// Each wave owns 32 hidden units, processed as TWO sequential 16-column
// passes so only acc[4] (32 VGPRs) is live during a GEMM pass — this keeps
// the wave under the 256-VGPR direct-address limit (v1 spilled to scratch).
// h0/h1 state is exchanged between waves via LDS (bf16, conflict-free padded
// stride); cell state c lives in VGPRs. GEMMs: v_wmma_f32_16x16x32_bf16.
// ---------------------------------------------------------------------------

typedef __attribute__((ext_vector_type(16))) __bf16 v16bf;
typedef __attribute__((ext_vector_type(8)))  float  v8f;

#define BB   256
#define TT   128
#define II   256
#define HH   512
#define G4   2048          // 4*H
#define MT   16            // batch tile per workgroup
#define HP   520           // padded LDS row stride for h (bf16): 1040B=260dw, 260%64=4 -> conflict-free
#define XP   264           // padded LDS row stride for x (bf16)

__device__ __forceinline__ unsigned short f32_to_bf16(float f) {
  unsigned int u = __float_as_uint(f);
  unsigned int r = u + 0x7FFFu + ((u >> 16) & 1u);
  return (unsigned short)(r >> 16);
}
__device__ __forceinline__ float bf16_to_f32(unsigned short v) {
  return __uint_as_float(((unsigned int)v) << 16);
}
__device__ __forceinline__ float sigf(float x) {
  return 1.0f / (1.0f + __expf(-x));
}
__device__ __forceinline__ float tanhfast(float x) {
  return 2.0f * sigf(2.0f * x) - 1.0f;
}

// One 16x32 bf16 WMMA operand fragment for this lane: two disjoint 8-elem
// (16B) runs at [k, k+8) and [k+16, k+24); k base already has +8 for hi lanes.
__device__ __forceinline__ v16bf ld_frag(const unsigned short* p) {
  union { uint4 q[2]; v16bf v; } u;
  u.q[0] = *(const uint4*)(p);
  u.q[1] = *(const uint4*)(p + 16);
  return u.v;
}

// acc[g] += A(16x32 from abase) * B(32x16 from W rows g*HH+nbase..+15),
// over KC k-chunks of 32. W row-major (4H x K) bf16; B lane n reads row
// (nbase + ln) contiguously in k. Only 4 accumulators live.
template <int KC>
__device__ __forceinline__ void gemm4(v8f acc[4],
                                      const unsigned short* abase, int astride,
                                      const unsigned short* __restrict__ W, int K,
                                      int nbase, int ln, int hi) {
#pragma unroll 2
  for (int kc = 0; kc < KC; ++kc) {
    v16bf a = ld_frag(abase + ln * astride + kc * 32 + hi * 8);
#pragma unroll
    for (int g = 0; g < 4; ++g) {
      const unsigned short* wp =
          W + (size_t)(g * HH + nbase + ln) * K + kc * 32 + hi * 8;
      v16bf b = ld_frag(wp);
      acc[g] = __builtin_amdgcn_wmma_f32_16x16x32_bf16(
          false, a, false, b, (short)0, acc[g], false, false);
    }
  }
}

// ---------------------------------------------------------------------------
// Kernel 1: fp32 -> bf16 weight conversion + bias pre-sum.
// ---------------------------------------------------------------------------
__global__ void lstm_convert_kernel(const float* __restrict__ Wih0,
                                    const float* __restrict__ Whh0,
                                    const float* __restrict__ Wih1,
                                    const float* __restrict__ Whh1,
                                    const float* __restrict__ bih0,
                                    const float* __restrict__ bhh0,
                                    const float* __restrict__ bih1,
                                    const float* __restrict__ bhh1,
                                    unsigned short* __restrict__ wbf,
                                    float* __restrict__ b0,
                                    float* __restrict__ b1) {
  const int tid = blockIdx.x * blockDim.x + threadIdx.x;
  const int stride = gridDim.x * blockDim.x;
  const int n0 = G4 * II;
  const int n1 = G4 * HH;
  unsigned short* p0 = wbf;
  unsigned short* p1 = p0 + n0;
  unsigned short* p2 = p1 + n1;
  unsigned short* p3 = p2 + n1;
  for (int i = tid; i < n0; i += stride) p0[i] = f32_to_bf16(Wih0[i]);
  for (int i = tid; i < n1; i += stride) p1[i] = f32_to_bf16(Whh0[i]);
  for (int i = tid; i < n1; i += stride) p2[i] = f32_to_bf16(Wih1[i]);
  for (int i = tid; i < n1; i += stride) p3[i] = f32_to_bf16(Whh1[i]);
  for (int i = tid; i < G4; i += stride) {
    b0[i] = bih0[i] + bhh0[i];
    b1[i] = bih1[i] + bhh1[i];
  }
}

// ---------------------------------------------------------------------------
// Kernel 2: recurrent LSTM. One WG per 16-row batch tile, 16 waves.
// ---------------------------------------------------------------------------
__global__ void __launch_bounds__(512, 1)
lstm_recurrent_kernel(const float* __restrict__ x,
                      const unsigned char* __restrict__ is_init,
                      const float* __restrict__ h_in,
                      const float* __restrict__ c_in,
                      const unsigned short* __restrict__ Wih0,
                      const unsigned short* __restrict__ Whh0,
                      const unsigned short* __restrict__ Wih1,
                      const unsigned short* __restrict__ Whh1,
                      const float* __restrict__ b0g,
                      const float* __restrict__ b1g,
                      float* __restrict__ out) {
  __shared__ __align__(16) unsigned short h0buf[MT * HP];
  __shared__ __align__(16) unsigned short h1buf[MT * HP];
  __shared__ __align__(16) unsigned short xbuf[MT * XP];
  __shared__ __align__(16) float b0s[G4];
  __shared__ __align__(16) float b1s[G4];
  __shared__ unsigned char maskbuf[MT];

  const int tid  = threadIdx.x;
  const int wave = tid >> 5;
  const int lane = tid & 31;
  const int ln   = lane & 15;
  const int hi   = lane >> 4;
  const int brow = blockIdx.x * MT;
  const int ncol0 = wave * 32;          // 32 hidden units per wave, 2 passes of 16

  for (int i = tid; i < G4; i += 512) { b0s[i] = b0g[i]; b1s[i] = b1g[i]; }

  for (int i = tid; i < MT * HH; i += 512) {
    int m = i >> 9, n = i & (HH - 1);
    h0buf[m * HP + n] = f32_to_bf16(h_in[(size_t)(brow + m) * 2 * HH + n]);
    h1buf[m * HP + n] = f32_to_bf16(h_in[(size_t)(brow + m) * 2 * HH + HH + n]);
  }

  // Cell state: C/D layout -> reg r, lane: [m = r + 8*hi, n = base + ln].
  v8f c0[2], c1[2];
#pragma unroll
  for (int hh = 0; hh < 2; ++hh) {
#pragma unroll
    for (int r = 0; r < 8; ++r) {
      int m = r + 8 * hi;
      int n = ncol0 + hh * 16 + ln;
      c0[hh][r] = c_in[(size_t)(brow + m) * 2 * HH + n];
      c1[hh][r] = c_in[(size_t)(brow + m) * 2 * HH + HH + n];
    }
  }

  for (int t = 0; t < TT; ++t) {
    // ---- stage x_t (bf16) + per-row reset mask ----
    if (tid < MT) maskbuf[tid] = is_init[(size_t)(brow + tid) * TT + t];
    for (int i = tid; i < MT * II; i += 512) {
      int m = i >> 8, n = i & (II - 1);
      xbuf[m * XP + n] = f32_to_bf16(x[((size_t)(brow + m) * TT + t) * II + n]);
    }
    __syncthreads();

    // ---- is_init: zero h rows in LDS + cell-state regs ----
    {
      int m = tid >> 5, j = tid & 31;
      if (maskbuf[m]) {
#pragma unroll
        for (int e = 0; e < 16; ++e) {
          h0buf[m * HP + j * 16 + e] = 0;
          h1buf[m * HP + j * 16 + e] = 0;
        }
      }
    }
#pragma unroll
    for (int r = 0; r < 8; ++r) {
      if (maskbuf[r + 8 * hi]) {
        c0[0][r] = 0.0f; c0[1][r] = 0.0f;
        c1[0][r] = 0.0f; c1[1][r] = 0.0f;
      }
    }
    __syncthreads();

    // ======================= layer 0 (two 16-col passes) =======================
    v8f h0new[2];
#pragma unroll 1
    for (int hh = 0; hh < 2; ++hh) {
      const int nb = ncol0 + hh * 16;
      v8f acc[4];
#pragma unroll
      for (int g = 0; g < 4; ++g) {
        float bv = b0s[g * HH + nb + ln];
#pragma unroll
        for (int r = 0; r < 8; ++r) acc[g][r] = bv;
      }
      gemm4<II / 32>(acc, xbuf,  XP, Wih0, II, nb, ln, hi);  // x_t @ W_ih0^T
      gemm4<HH / 32>(acc, h0buf, HP, Whh0, HH, nb, ln, hi);  // h0  @ W_hh0^T
#pragma unroll
      for (int r = 0; r < 8; ++r) {
        float iv = sigf(acc[0][r]);
        float fv = sigf(acc[1][r]);
        float gv = tanhfast(acc[2][r]);
        float ov = sigf(acc[3][r]);
        float cn = fv * c0[hh][r] + iv * gv;
        c0[hh][r] = cn;
        h0new[hh][r] = ov * tanhfast(cn);
      }
    }

    __syncthreads();  // all waves done reading h0buf
#pragma unroll
    for (int hh = 0; hh < 2; ++hh)
#pragma unroll
      for (int r = 0; r < 8; ++r) {
        int m = r + 8 * hi, n = ncol0 + hh * 16 + ln;
        h0buf[m * HP + n] = f32_to_bf16(h0new[hh][r]);
      }
    __syncthreads();  // h0buf ready for layer 1

    // ======================= layer 1 (two 16-col passes) =======================
    v8f h1new[2];
#pragma unroll 1
    for (int hh = 0; hh < 2; ++hh) {
      const int nb = ncol0 + hh * 16;
      v8f acc[4];
#pragma unroll
      for (int g = 0; g < 4; ++g) {
        float bv = b1s[g * HH + nb + ln];
#pragma unroll
        for (int r = 0; r < 8; ++r) acc[g][r] = bv;
      }
      gemm4<HH / 32>(acc, h0buf, HP, Wih1, HH, nb, ln, hi);  // h0new @ W_ih1^T
      gemm4<HH / 32>(acc, h1buf, HP, Whh1, HH, nb, ln, hi);  // h1    @ W_hh1^T
#pragma unroll
      for (int r = 0; r < 8; ++r) {
        float iv = sigf(acc[0][r]);
        float fv = sigf(acc[1][r]);
        float gv = tanhfast(acc[2][r]);
        float ov = sigf(acc[3][r]);
        float cn = fv * c1[hh][r] + iv * gv;
        c1[hh][r] = cn;
        h1new[hh][r] = ov * tanhfast(cn);
      }
    }

    __syncthreads();  // all waves done reading h1buf
#pragma unroll
    for (int hh = 0; hh < 2; ++hh)
#pragma unroll
      for (int r = 0; r < 8; ++r) {
        int m = r + 8 * hi, n = ncol0 + hh * 16 + ln;
        h1buf[m * HP + n] = f32_to_bf16(h1new[hh][r]);
        out[((size_t)(brow + m) * TT + t) * HH + n] = h1new[hh][r];
      }
    __syncthreads();
  }

  // ---- final states: h_n (from LDS), c_n (from regs) ----
  float* hn = out + (size_t)BB * TT * HH;
  float* cn = hn + (size_t)BB * 2 * HH;
#pragma unroll
  for (int hh = 0; hh < 2; ++hh)
#pragma unroll
    for (int r = 0; r < 8; ++r) {
      int m = r + 8 * hi, n = ncol0 + hh * 16 + ln;
      cn[(size_t)(brow + m) * 2 * HH + n]      = c0[hh][r];
      cn[(size_t)(brow + m) * 2 * HH + HH + n] = c1[hh][r];
    }
  for (int i = tid; i < MT * HH; i += 512) {
    int m = i >> 9, n = i & (HH - 1);
    hn[(size_t)(brow + m) * 2 * HH + n]      = bf16_to_f32(h0buf[m * HP + n]);
    hn[(size_t)(brow + m) * 2 * HH + HH + n] = bf16_to_f32(h1buf[m * HP + n]);
  }
}

// ---------------------------------------------------------------------------
extern "C" void kernel_launch(void* const* d_in, const int* in_sizes, int n_in,
                              void* d_out, int out_size, void* d_ws, size_t ws_size,
                              hipStream_t stream) {
  const float*         x       = (const float*)d_in[0];
  const unsigned char* is_init = (const unsigned char*)d_in[1];
  const float*         h       = (const float*)d_in[2];
  const float*         c       = (const float*)d_in[3];
  const float*         Wih0    = (const float*)d_in[4];
  const float*         Whh0    = (const float*)d_in[5];
  const float*         bih0    = (const float*)d_in[6];
  const float*         bhh0    = (const float*)d_in[7];
  const float*         Wih1    = (const float*)d_in[8];
  const float*         Whh1    = (const float*)d_in[9];
  const float*         bih1    = (const float*)d_in[10];
  const float*         bhh1    = (const float*)d_in[11];
  float* out = (float*)d_out;

  unsigned short* wbf = (unsigned short*)d_ws;
  const size_t n0 = (size_t)G4 * II;
  const size_t n1 = (size_t)G4 * HH;
  unsigned short* wih0 = wbf;
  unsigned short* whh0 = wih0 + n0;
  unsigned short* wih1 = whh0 + n1;
  unsigned short* whh1 = wih1 + n1;
  float* b0 = (float*)(whh1 + n1);
  float* b1 = b0 + G4;

  lstm_convert_kernel<<<512, 256, 0, stream>>>(Wih0, Whh0, Wih1, Whh1,
                                               bih0, bhh0, bih1, bhh1,
                                               wbf, b0, b1);

  lstm_recurrent_kernel<<<BB / MT, 512, 0, stream>>>(
      x, is_init, h, c, wih0, whh0, wih1, whh1, b0, b1, out);
}
